// ODE_71743133713072
// MI455X (gfx1250) — matrix operand order repaired and checked by reading, not compiled, not run on hardware
//
#include <hip/hip_runtime.h>
#include <hip/hip_bf16.h>
#include <math.h>

// Problem constants (B, T_OBS, K_PRED, D, U) = (1024, 100, 20, 256, 256)
#define DD 256          // feature dim (D == U == 256)
#define MT 64           // rows per workgroup tile
#define PITCH 264       // bf16 elems per LDS row: 528B, 16B-aligned, bank-skewed (+4 banks/row)

#define GA __attribute__((address_space(1)))   // global address space -> global_load/store

typedef __attribute__((ext_vector_type(16))) __bf16 v16bf;
typedef __attribute__((ext_vector_type(8)))  __bf16 v8bf;
typedef __attribute__((ext_vector_type(4)))  __bf16 v4bf;
typedef __attribute__((ext_vector_type(8)))  float  v8f;
typedef __attribute__((ext_vector_type(4)))  float  v4f;

union ABReg { v16bf v; v8bf h[2]; };

// gfx1250 has a hardware TANH transcendental (TRANS32). Use it if the
// toolchain exposes the builtin; otherwise fall back to libm tanhf.
__device__ __forceinline__ float dev_tanh(float x) {
#if __has_builtin(__builtin_amdgcn_tanhf)
  return __builtin_amdgcn_tanhf(x);
#elif __has_builtin(__builtin_amdgcn_tanh_f32)
  return __builtin_amdgcn_tanh_f32(x);
#else
  return tanhf(x);
#endif
}

// One Euler step on MT rows: out_row = in_row + (W3(tanh(W2(tanh(W1 x + b1)) + b2)) + b3) * dt
__global__ __launch_bounds__(256) void euler_step_kernel(
    const float* __restrict__ in_ptr, long in_bstride, long in_xstride,
    float* __restrict__ out_ptr, long out_bstride, long out_xstride,
    const float* __restrict__ ts, int t_off,
    const __bf16* __restrict__ wt1, const __bf16* __restrict__ wt2,
    const __bf16* __restrict__ wt3,
    const float* __restrict__ bias1, const float* __restrict__ bias2,
    const float* __restrict__ bias3)
{
  __shared__ __align__(16) __bf16 act[MT * PITCH];

  const int tid   = threadIdx.x;
  const int lane  = tid & 31;
  const int lmod  = lane & 15;
  const int lhalf = lane >> 4;
  // wave-uniform tile coordinates, forced into SGPRs
  const int wave  = __builtin_amdgcn_readfirstlane(tid >> 5);   // 8 waves (wave32)
  const int mrow  = wave >> 1;       // 0..3 : which 16-row band
  const int nblk  = (wave & 1) * 8;  // 8 N-tiles per wave

  const int b0 = blockIdx.y * MT;
  const GA float* inb  = (const GA float*)in_ptr  + (long)blockIdx.x * in_xstride;
  GA float*       outb = (GA float*)out_ptr       + (long)blockIdx.x * out_xstride;
  const GA float* tsg  = (const GA float*)ts;
  const float dt = tsg[t_off + blockIdx.x + 1] - tsg[t_off + blockIdx.x];

  // ---- stage input rows (f32 -> bf16) into LDS, vectorized ----
  {
    const int r0 = tid >> 6;         // 0..3
    const int c0 = (tid & 63) * 4;   // 0..252
#pragma unroll
    for (int it = 0; it < MT / 4; ++it) {
      const int row = it * 4 + r0;
      const v4f v = *(const GA v4f*)(inb + (long)(b0 + row) * in_bstride + c0);
      v4bf p;
      p.x = (__bf16)v.x; p.y = (__bf16)v.y; p.z = (__bf16)v.z; p.w = (__bf16)v.w;
      *(v4bf*)(&act[row * PITCH + c0]) = p;   // ds_store_b64
    }
  }
  __syncthreads();

  const __bf16* wts[3] = { wt1, wt2, wt3 };
  const float*  bss[3] = { bias1, bias2, bias3 };

  // Per-lane byte offset into a weight tile row (same for every nt/kc):
  //   Wt[n][k] bf16, n = tileN*16 + lmod, k chunk base 16*lhalf
  const int wlaneoff = lmod * (DD * 2) + lhalf * 32;

  for (int layer = 0; layer < 3; ++layer) {
    // SGPR base: weights of this layer, advanced to this wave's N-block
    const GA char* wbase =
        (const GA char*)wts[layer] + (long)nblk * (16 * DD * 2);
    const GA float* bp = (const GA float*)bss[layer];

    v8f acc[8] = {};

    const __bf16* arow = &act[(mrow * 16 + lmod) * PITCH];
#pragma unroll
    for (int kc = 0; kc < 8; ++kc) {           // K = 256 in chunks of 32
      // A (16x32 bf16): lane half h holds K 16h..16h+15 contiguous (matches B)
      ABReg a;
      const __bf16* ap = arow + kc * 32 + 16 * lhalf;
      a.h[0] = *(const v8bf*)(ap);             // ds_load_b128
      a.h[1] = *(const v8bf*)(ap + 8);         // ds_load_b128
#pragma unroll
      for (int nt = 0; nt < 8; ++nt) {
        // B (32x16 bf16): lane = col n, K = 16*lhalf..+15 contiguous in Wt[n][k]
        // address = one per-lane address + immediate (nt,kc) offset
        const GA v8bf* wp =
            (const GA v8bf*)(wbase + wlaneoff + nt * (16 * DD * 2) + kc * 64);
        ABReg b;
        b.h[0] = wp[0];                        // global_load_b128 offset:imm
        b.h[1] = wp[1];                        // global_load_b128 offset:imm+16
        acc[nt] = __builtin_amdgcn_wmma_f32_16x16x32_bf16(
            false, a.v, false, b.v, (short)0, acc[nt], false, false);
      }
    }

    const int mbase = mrow * 16 + 8 * lhalf;   // C/D layout: lane half -> row offset 0/8
    if (layer < 2) {
      __syncthreads();                         // all LDS reads of this layer's input done
#pragma unroll
      for (int nt = 0; nt < 8; ++nt) {
        const int col  = (nblk + nt) * 16 + lmod;
        const float bv = bp[col];
#pragma unroll
        for (int j = 0; j < 8; ++j)
          act[(mbase + j) * PITCH + col] = (__bf16)dev_tanh(acc[nt][j] + bv);
      }
      __syncthreads();
    } else {
      // Fused epilogue: out = x + (h@W3 + b3) * dt
#pragma unroll
      for (int nt = 0; nt < 8; ++nt) {
        const int col  = (nblk + nt) * 16 + lmod;
        const float bv = bp[col];
#pragma unroll
        for (int j = 0; j < 8; ++j) {
          const long brow = (long)(b0 + mbase + j);
          const float x = inb[brow * in_bstride + col];
          outb[brow * out_bstride + col] = fmaf(acc[nt][j] + bv, dt, x);
        }
      }
    }
  }
}

// Convert W1/W2/W3 (f32, row-major [K][N]) -> bf16 transposed Wt[N][K] in workspace.
__global__ void convert_weights_kernel(const float* __restrict__ W1,
                                       const float* __restrict__ W2,
                                       const float* __restrict__ W3,
                                       __bf16* __restrict__ wt)
{
  const int id  = blockIdx.x * 256 + threadIdx.x;   // 3*65536 elems
  const int mat = id >> 16;
  const int rem = id & 65535;
  const int k = rem >> 8, n = rem & 255;
  const float* W = (mat == 0) ? W1 : (mat == 1) ? W2 : W3;
  ((GA __bf16*)wt)[mat * 65536 + n * DD + k] = (__bf16)((const GA float*)W)[k * DD + n];
}

// out[:, 0, :] = latents[:, 0, :]
__global__ void copy_first_kernel(const float* __restrict__ lat, float* __restrict__ out)
{
  const long b = blockIdx.x;
  ((GA float*)out)[b * (120L * DD) + threadIdx.x] =
      ((const GA float*)lat)[b * (100L * DD) + threadIdx.x];
}

extern "C" void kernel_launch(void* const* d_in, const int* in_sizes, int n_in,
                              void* d_out, int out_size, void* d_ws, size_t ws_size,
                              hipStream_t stream) {
  (void)in_sizes; (void)n_in; (void)out_size; (void)ws_size;
  const float* latents = (const float*)d_in[0];
  const float* ts      = (const float*)d_in[1];
  const float* W1 = (const float*)d_in[3]; const float* b1 = (const float*)d_in[4];
  const float* W2 = (const float*)d_in[5]; const float* b2 = (const float*)d_in[6];
  const float* W3 = (const float*)d_in[7]; const float* b3 = (const float*)d_in[8];
  float* out = (float*)d_out;

  __bf16* wt  = reinterpret_cast<__bf16*>(d_ws);   // 3 * 256*256 bf16 = 384 KB
  __bf16* wt1 = wt;
  __bf16* wt2 = wt + 65536;
  __bf16* wt3 = wt + 131072;

  convert_weights_kernel<<<768, 256, 0, stream>>>(W1, W2, W3, wt);
  copy_first_kernel<<<1024, 256, 0, stream>>>(latents, out);

  // Phase 1: 99 teacher-forced steps, fully parallel. out[:, k+1] = x_k + f(x_k)*dt
  euler_step_kernel<<<dim3(99, 1024 / MT), 256, 0, stream>>>(
      latents, 100L * DD, (long)DD,
      out + DD, 120L * DD, (long)DD,
      ts, 0, wt1, wt2, wt3, b1, b2, b3);

  // Phase 2: 20 sequential prediction steps (autoregressive rollout).
  for (int i = 0; i < 20; ++i) {
    const float* in_p = (i == 0) ? (latents + 99L * DD) : (out + (99L + i) * DD);
    const long  in_bs = (i == 0) ? 100L * DD : 120L * DD;
    euler_step_kernel<<<dim3(1, 1024 / MT), 256, 0, stream>>>(
        in_p, in_bs, 0L,
        out + (100L + i) * DD, 120L * DD, 0L,
        ts, 99 + i, wt1, wt2, wt3, b1, b2, b3);
  }
}
